// BSFFL_14585708937689
// MI455X (gfx1250) — matrix operand order repaired and checked by reading, not compiled, not run on hardware
//
#include <hip/hip_runtime.h>
#include <math.h>

#define D_MODEL 256
#define D_FF    1024
#define N_B     4
#define NTOK    65536     // 32*2048
#define TILE_M  64
#define MAX_TILES 1024    // NTOK / TILE_M

typedef __attribute__((ext_vector_type(16))) __bf16       v16bf;
typedef __attribute__((ext_vector_type(8)))  float        v8f;
typedef __attribute__((ext_vector_type(4)))  unsigned int u32x4;

union Frag { v16bf v; u32x4 q[2]; };

__device__ __forceinline__ unsigned short f2bf(float f) {
    unsigned int u = __float_as_uint(f);
    u += 0x7FFFu + ((u >> 16) & 1u);      // round-to-nearest-even
    return (unsigned short)(u >> 16);
}

// ---------------- pass 1: zero output ----------------
__global__ void k_zero(float4* __restrict__ out, int n4) {
    int i = blockIdx.x * blockDim.x + threadIdx.x;
    if (i < n4) out[i] = make_float4(0.f, 0.f, 0.f, 0.f);
}

// ---------------- pass 2: weights -> bf16, zero counters ----------------
__global__ void k_prep(const float* __restrict__ w1, const float* __restrict__ w2,
                       unsigned short* __restrict__ w1b, unsigned short* __restrict__ w2b,
                       int* __restrict__ counts) {
    int i = blockIdx.x * blockDim.x + threadIdx.x;
    const int n = N_B * D_FF * D_MODEL;
    if (i < n) {
        w1b[i] = f2bf(w1[i]);
        w2b[i] = f2bf(w2[i]);
    }
    if (i < N_B) counts[i] = 0;
}

// ---------------- pass 3: route tokens by branch ----------------
__global__ void k_route(const int* __restrict__ b_seq, int* __restrict__ counts,
                        int* __restrict__ lists) {
    int t = blockIdx.x * blockDim.x + threadIdx.x;
    if (t < NTOK) {
        int b = b_seq[t];
        if (b > 0) {
            int p = atomicAdd(&counts[b - 1], 1);
            lists[(b - 1) * NTOK + p] = t;
        }
    }
}

// ---------------- pass 4: fused routed FFN + LN ----------------
__global__ void __launch_bounds__(256)
k_ffn(const float* __restrict__ x,
      const unsigned short* __restrict__ w1b,
      const unsigned short* __restrict__ w2b,
      const float* __restrict__ b1,
      const float* __restrict__ b2,
      const float* __restrict__ gamma,
      const float* __restrict__ beta,
      const int* __restrict__ counts,
      const int* __restrict__ lists,
      float* __restrict__ out) {

    const int branch = blockIdx.x >> 10;
    const int tile   = blockIdx.x & (MAX_TILES - 1);
    const int cnt    = counts[branch];
    if (tile * TILE_M >= cnt) return;

    __shared__ unsigned short sA[TILE_M * D_MODEL];   // 32 KB gathered x (bf16)
    __shared__ unsigned short sH[2][TILE_M * 64];     // 2x8 KB h-chunk (bf16)
    __shared__ int   sTok[TILE_M];
    __shared__ int   sValid[TILE_M];
    __shared__ float sS1[TILE_M][2];
    __shared__ float sS2[TILE_M][2];

    const int tid = threadIdx.x;
    const int* list = lists + branch * NTOK;

    if (tid < TILE_M) {
        int g  = tile * TILE_M + tid;
        int ok = g < cnt;
        sTok[tid]   = list[ok ? g : (cnt - 1)];
        sValid[tid] = ok;
    }
    __syncthreads();

    // gather x rows, convert to bf16 into LDS (row-major [64][256])
    {
        int r0 = tid >> 6;            // 0..3
        int c4 = (tid & 63) * 4;      // column base (multiple of 4)
        for (int rr = 0; rr < TILE_M; rr += 4) {
            int row = rr + r0;
            const float4 xv = *(const float4*)(x + (size_t)sTok[row] * D_MODEL + c4);
            unsigned long long pk =  (unsigned long long)f2bf(xv.x)
                                  | ((unsigned long long)f2bf(xv.y) << 16)
                                  | ((unsigned long long)f2bf(xv.z) << 32)
                                  | ((unsigned long long)f2bf(xv.w) << 48);
            *(unsigned long long*)(&sA[row * D_MODEL + c4]) = pk;
        }
    }
    __syncthreads();

    const int wave = tid >> 5;
    const int lane = tid & 31;
    const int lh   = lane >> 4;    // lane half (0/1)
    const int l16  = lane & 15;
    const int mt   = wave >> 1;    // M-tile 0..3 (16 rows each)
    const int fh   = wave & 1;     // half of the 64-wide f-chunk / half of y cols

    // ---- hoist GEMM1 A-fragments (loop-invariant: x tile, K=256) ----
    Frag aX[8];
    {
        const unsigned short* ab = &sA[(mt * 16 + l16) * D_MODEL + lh * 8];
        #pragma unroll
        for (int k = 0; k < 8; k++) {
            aX[k].q[0] = *(const u32x4*)(ab + k * 32);
            aX[k].q[1] = *(const u32x4*)(ab + k * 32 + 16);
        }
    }

    v8f accY[8];
    #pragma unroll
    for (int j = 0; j < 8; j++) accY[j] = {};

    const float*          b1p = b1  + branch * D_FF;
    const unsigned short* w1p = w1b + (size_t)branch * D_FF * D_MODEL;
    const unsigned short* w2p = w2b + (size_t)branch * D_MODEL * D_FF;

    int buf = 0;
    for (int kf = 0; kf < D_FF; kf += 64, buf ^= 1) {
        // ---- GEMM1: two 16-col subtiles of this 64-wide f-chunk ----
        const int f0 = kf + (fh * 2 + 0) * 16 + l16;
        const int f1 = kf + (fh * 2 + 1) * 16 + l16;
        const unsigned short* wb0 = w1p + (size_t)f0 * D_MODEL + lh * 8;
        const unsigned short* wb1 = w1p + (size_t)f1 * D_MODEL + lh * 8;
        v8f acc0 = {}, acc1 = {};
        #pragma unroll
        for (int k = 0; k < 8; k++) {
            Frag b0, b1f;
            b0.q[0]  = *(const u32x4*)(wb0 + k * 32);
            b0.q[1]  = *(const u32x4*)(wb0 + k * 32 + 16);
            b1f.q[0] = *(const u32x4*)(wb1 + k * 32);
            b1f.q[1] = *(const u32x4*)(wb1 + k * 32 + 16);
            acc0 = __builtin_amdgcn_wmma_f32_16x16x32_bf16(
                       false, aX[k].v, false, b0.v,  (short)0, acc0, false, false);
            acc1 = __builtin_amdgcn_wmma_f32_16x16x32_bf16(
                       false, aX[k].v, false, b1f.v, (short)0, acc1, false, false);
        }

        // bias + ELU + pack into LDS h-chunk (row-major [64][64])
        {
            const float bb0 = b1p[f0];
            const float bb1 = b1p[f1];
            const int hrow0 = mt * 16 + lh * 8;
            unsigned short* hdst = &sH[buf][0];
            #pragma unroll
            for (int r = 0; r < 8; r++) {
                float v0 = acc0[r] + bb0;
                float v1 = acc1[r] + bb1;
                v0 = v0 > 0.0f ? v0 : (__expf(v0) - 1.0f);
                v1 = v1 > 0.0f ? v1 : (__expf(v1) - 1.0f);
                unsigned short* row = hdst + (hrow0 + r) * 64;
                row[(fh * 2 + 0) * 16 + l16] = f2bf(v0);
                row[(fh * 2 + 1) * 16 + l16] = f2bf(v1);
            }
        }
        __syncthreads();

        // ---- GEMM2 partial: y += h_chunk * w2_chunk^T (2 k-substeps) ----
        #pragma unroll
        for (int ks = 0; ks < 2; ks++) {
            Frag ah;
            const unsigned short* hb = &sH[buf][(mt * 16 + l16) * 64 + ks * 32 + lh * 8];
            ah.q[0] = *(const u32x4*)(hb);
            ah.q[1] = *(const u32x4*)(hb + 16);
            #pragma unroll
            for (int j = 0; j < 8; j++) {
                const int col = (fh * 8 + j) * 16 + l16;
                const unsigned short* w2l = w2p + (size_t)col * D_FF + kf + ks * 32 + lh * 8;
                Frag b;
                b.q[0] = *(const u32x4*)(w2l);
                b.q[1] = *(const u32x4*)(w2l + 16);
                accY[j] = __builtin_amdgcn_wmma_f32_16x16x32_bf16(
                              false, ah.v, false, b.v, (short)0, accY[j], false, false);
            }
        }
    }

    // ---- epilogue: bias2, LayerNorm, scatter ----
    float g8[8], be8[8], b28[8];
    #pragma unroll
    for (int j = 0; j < 8; j++) {
        const int col = (fh * 8 + j) * 16 + l16;
        b28[j] = b2[branch * D_MODEL + col];
        g8[j]  = gamma[branch * D_MODEL + col];
        be8[j] = beta[branch * D_MODEL + col];
    }

    float s1r[8], s2r[8];
    #pragma unroll
    for (int r = 0; r < 8; r++) {
        float s1 = 0.f, s2 = 0.f;
        #pragma unroll
        for (int j = 0; j < 8; j++) {
            float v = accY[j][r] + b28[j];
            accY[j][r] = v;
            s1 += v; s2 += v * v;
        }
        s1r[r] = s1; s2r[r] = s2;
    }
    #pragma unroll
    for (int m = 1; m < 16; m <<= 1) {
        #pragma unroll
        for (int r = 0; r < 8; r++) {
            s1r[r] += __shfl_xor(s1r[r], m, 32);
            s2r[r] += __shfl_xor(s2r[r], m, 32);
        }
    }
    if (l16 == 0) {
        #pragma unroll
        for (int r = 0; r < 8; r++) {
            const int row = mt * 16 + lh * 8 + r;
            sS1[row][fh] = s1r[r];
            sS2[row][fh] = s2r[r];
        }
    }
    __syncthreads();

    #pragma unroll
    for (int r = 0; r < 8; r++) {
        const int row = mt * 16 + lh * 8 + r;
        const float t1 = sS1[row][0] + sS1[row][1];
        const float t2 = sS2[row][0] + sS2[row][1];
        const float mu  = t1 * (1.0f / 256.0f);
        const float var = t2 * (1.0f / 256.0f) - mu * mu;
        const float rs  = rsqrtf(var + 1e-12f);
        if (sValid[row]) {
            float* orow = out + (size_t)sTok[row] * D_MODEL;
            #pragma unroll
            for (int j = 0; j < 8; j++) {
                const int col = (fh * 8 + j) * 16 + l16;
                orow[col] = (accY[j][r] - mu) * rs * g8[j] + be8[j];
            }
        }
    }
}

extern "C" void kernel_launch(void* const* d_in, const int* in_sizes, int n_in,
                              void* d_out, int out_size, void* d_ws, size_t ws_size,
                              hipStream_t stream) {
    const float* x     = (const float*)d_in[0];
    const int*   b_seq = (const int*)d_in[1];
    const float* w1    = (const float*)d_in[2];
    const float* b1    = (const float*)d_in[3];
    const float* w2    = (const float*)d_in[4];
    const float* b2    = (const float*)d_in[5];
    const float* gam   = (const float*)d_in[6];
    const float* bet   = (const float*)d_in[7];
    float* out = (float*)d_out;

    char* ws = (char*)d_ws;
    const size_t wsz = (size_t)N_B * D_FF * D_MODEL;   // elements per weight tensor
    unsigned short* w1b   = (unsigned short*)ws;                         // 2 MB
    unsigned short* w2b   = (unsigned short*)(ws + wsz * 2);             // 2 MB
    int*            cnts  = (int*)(ws + wsz * 4);                        // 16 B
    int*            lists = (int*)(ws + wsz * 4 + 256);                  // 1 MB

    const int n4 = out_size / 4;
    k_zero <<<(n4 + 255) / 256, 256, 0, stream>>>((float4*)out, n4);

    const int nw = (int)wsz;
    k_prep <<<(nw + 255) / 256, 256, 0, stream>>>(w1, w2, w1b, w2b, cnts);

    k_route<<<NTOK / 256, 256, 0, stream>>>(b_seq, cnts, lists);

    k_ffn  <<<N_B * MAX_TILES, 256, 0, stream>>>(x, w1b, w2b, b1, b2, gam, bet,
                                                 cnts, lists, out);
    (void)in_sizes; (void)n_in; (void)ws_size;
}
